// LazyAttention_79388175499697
// MI455X (gfx1250) — matrix operand order repaired and checked by reading, not compiled, not run on hardware
//
#include <hip/hip_runtime.h>
#include <hip/hip_bf16.h>
#include <hip/hip_fp16.h>

typedef __attribute__((ext_vector_type(16))) _Float16 v16h;
typedef __attribute__((ext_vector_type(8)))  float    v8f;

#define H_NUM   16
#define WIN     512
#define HD      64
#define DM      1024
#define LSEQ    2048
#define BLROWS  4096   // B*L

static __device__ __forceinline__ v8f wmma16(v16h a, v16h b, v8f c) {
    return __builtin_amdgcn_wmma_f32_16x16x32_f16(false, a, false, b, (short)0, c, false, false);
}

// Async global->LDS copy of 16 bytes per lane (gfx1250 ASYNCcnt path).
// saddr: uniform 64-bit base (SGPR pair), voff: per-lane byte offset,
// lds: destination in shared memory (low 32 bits of generic ptr = LDS addr).
static __device__ __forceinline__ void async_copy_b128(void* lds, const void* saddr, unsigned voff) {
    unsigned l = (unsigned)(size_t)lds;
    asm volatile("global_load_async_to_lds_b128 %0, %1, %2"
                 :: "v"(l), "v"(voff), "s"(saddr) : "memory");
}
static __device__ __forceinline__ void wait_async0() {
    asm volatile("s_wait_asynccnt 0x0" ::: "memory");
}

// ---------------------------------------------------------------------------
// Kernel 1: Q/K/V projections (x @ W^T) with fused RoPE + q-scale, f16 out.
// grid (BLROWS/128, H), 256 threads (8 waves, 16 rows each, 64 cols = head).
// LDS tiles staged as f16 (converted once at staging time).
// ---------------------------------------------------------------------------
__global__ __launch_bounds__(256) void proj_rope_kernel(
    const float* __restrict__ x,  const float* __restrict__ Wq,
    const float* __restrict__ Wk, const float* __restrict__ Wv,
    _Float16* __restrict__ Qh, _Float16* __restrict__ Kh, _Float16* __restrict__ Vh)
{
    const int h        = blockIdx.y;
    const int rowBase0 = blockIdx.x * 128;       // row within [0, B*L)
    const int tid      = threadIdx.x;
    const int lane     = tid & 31;
    const int wave     = tid >> 5;
    const int lane15   = lane & 15;
    const int laneHi   = lane >> 4;

    __shared__ _Float16 Xs[128][40];             // x tile  (rows x 32 k), f16
    __shared__ _Float16 Ws[3][64][40];           // W tiles (64 n x 32 k), f16

    v8f acc[3][4];
    {
        v8f z = {};
        #pragma unroll
        for (int m = 0; m < 3; ++m)
            #pragma unroll
            for (int nt = 0; nt < 4; ++nt) acc[m][nt] = z;
    }

    for (int kk0 = 0; kk0 < DM; kk0 += 32) {
        __syncthreads();
        // stage x tile: 128x32, cvt f32->f16 during staging
        #pragma unroll
        for (int j = 0; j < 4; ++j) {
            int idx = tid + j * 256;
            int row = idx >> 3, c4 = idx & 7;
            float4 v = *(const float4*)(x + (size_t)(rowBase0 + row) * DM + kk0 + c4 * 4);
            Xs[row][c4*4+0] = (_Float16)v.x; Xs[row][c4*4+1] = (_Float16)v.y;
            Xs[row][c4*4+2] = (_Float16)v.z; Xs[row][c4*4+3] = (_Float16)v.w;
        }
        // stage Wq/Wk/Wv tiles: 3 x 64x32
        #pragma unroll
        for (int j = 0; j < 6; ++j) {
            int idx = tid + j * 256;
            int mat = idx >> 9;
            int rem = idx & 511;
            int wr = rem >> 3, c4 = rem & 7;
            const float* Wp = (mat == 0) ? Wq : ((mat == 1) ? Wk : Wv);
            float4 v = *(const float4*)(Wp + (size_t)(h * HD + wr) * DM + kk0 + c4 * 4);
            Ws[mat][wr][c4*4+0] = (_Float16)v.x; Ws[mat][wr][c4*4+1] = (_Float16)v.y;
            Ws[mat][wr][c4*4+2] = (_Float16)v.z; Ws[mat][wr][c4*4+3] = (_Float16)v.w;
        }
        __syncthreads();

        // A fragment (16x32 f16), contiguous k = 16*laneHi + s; adjacent halfs -> b32 ds loads
        v16h af;
        int arow = wave * 16 + lane15;
        #pragma unroll
        for (int r = 0; r < 8; ++r) {
            int kk = 2 * r + (laneHi ? 16 : 0);
            af[2*r]   = Xs[arow][kk];
            af[2*r+1] = Xs[arow][kk+1];
        }
        #pragma unroll
        for (int mat = 0; mat < 3; ++mat) {
            #pragma unroll
            for (int nt = 0; nt < 4; ++nt) {
                v16h bf;
                int n = nt * 16 + lane15;
                #pragma unroll
                for (int r = 0; r < 8; ++r) {
                    int kk = 2 * r + (laneHi ? 16 : 0);
                    bf[2*r]   = Ws[mat][n][kk];
                    bf[2*r+1] = Ws[mat][n][kk+1];
                }
                acc[mat][nt] = wmma16(af, bf, acc[mat][nt]);
            }
        }
    }

    // ---- epilogue: RoPE (lane-local: d pairs with d^32 => nt^2), q *= 1/8 ----
    const int mrowbase = rowBase0 + wave * 16;
    float qn[4][8], kn[4][8];
    #pragma unroll
    for (int nt = 0; nt < 4; ++nt) {
        int d = nt * 16 + lane15;
        int i = d & 31;
        float invf = powf(10000.0f, -(float)i * (1.0f / 32.0f));
        #pragma unroll
        for (int r = 0; r < 8; ++r) {
            int l = (mrowbase + r + laneHi * 8) & (LSEQ - 1);
            float ang = (float)l * invf;
            float c = cosf(ang), s = sinf(ang);
            float qv = acc[0][nt][r],   kv = acc[1][nt][r];
            float qp = acc[0][nt ^ 2][r], kp = acc[1][nt ^ 2][r];
            float rq = (d < 32) ? -qp : qp;
            float rk = (d < 32) ? -kp : kp;
            qn[nt][r] = (qv * c + rq * s) * 0.125f;   // fold scale hd^-0.5
            kn[nt][r] = (kv * c + rk * s);
        }
    }
    const int b = rowBase0 >> 11;                 // 2048 rows per batch
    const size_t base = ((size_t)(b * H_NUM + h)) * LSEQ * HD;
    #pragma unroll
    for (int nt = 0; nt < 4; ++nt) {
        int d = nt * 16 + lane15;
        #pragma unroll
        for (int r = 0; r < 8; ++r) {
            int l = (mrowbase + r + laneHi * 8) & (LSEQ - 1);
            size_t off = base + (size_t)l * HD + d;
            Qh[off] = (_Float16)qn[nt][r];
            Kh[off] = (_Float16)kn[nt][r];
            Vh[off] = (_Float16)acc[2][nt][r];
        }
    }
}

// ---------------------------------------------------------------------------
// Kernel 2: two-pass causal attention with window bias + elastic softmax.
// grid (L/128, H, B), 256 threads. K/V tiles staged with async-to-LDS B128.
// ---------------------------------------------------------------------------
__global__ __launch_bounds__(256) void attn_kernel(
    const _Float16* __restrict__ Qh, const _Float16* __restrict__ Kh,
    const _Float16* __restrict__ Vh, const float* __restrict__ biases,
    const float* __restrict__ tau, _Float16* __restrict__ Obuf)
{
    const int bIdx   = blockIdx.z;
    const int h      = blockIdx.y;
    const int rowBase = blockIdx.x * 128;
    const int tid    = threadIdx.x;
    const int lane   = tid & 31;
    const int wave   = tid >> 5;
    const int lane15 = lane & 15;
    const int laneHi = lane >> 4;

    __shared__ _Float16 Kt[32][80];       // padded rows (160B) for bank spread
    __shared__ _Float16 Vt[32][80];
    __shared__ _Float16 Pt[8][16][40];    // per-wave P tile, C->A relayout
    __shared__ float    Bh[WIN + 1];

    for (int i = tid; i < WIN + 1; i += 256) Bh[i] = biases[h * (WIN + 1) + i];
    const float tauh = tau[h];
    const size_t hbase = ((size_t)(bIdx * H_NUM + h)) * LSEQ * HD;

    // per-lane async staging coordinates (one B128 per lane per tile)
    const int jj = tid >> 3;
    const int dk = (tid & 7) * 8;
    const unsigned voff = (unsigned)((jj * HD + dk) * sizeof(_Float16));

    // persistent Q A-fragments (hd=64 -> two 16x32 k-steps)
    v16h qa0, qa1;
    {
        const _Float16* qp = Qh + hbase + (size_t)(rowBase + wave * 16 + lane15) * HD;
        #pragma unroll
        for (int r = 0; r < 8; ++r) {
            int kk = 2 * r + (laneHi ? 16 : 0);
            qa0[2*r] = qp[kk];      qa0[2*r+1] = qp[kk + 1];
            qa1[2*r] = qp[32 + kk]; qa1[2*r+1] = qp[32 + kk + 1];
        }
    }

    int rl[8];
    float mrow[8], zrow[8];
    #pragma unroll
    for (int r = 0; r < 8; ++r) {
        rl[r] = rowBase + wave * 16 + r + laneHi * 8;
        mrow[r] = -1e30f; zrow[r] = 0.0f;
    }

    const int kmax = rowBase + 128;       // causal extent of this row block
    __syncthreads();                      // Bh ready

    // ---------------- pass 1: row max + Z ----------------
    for (int kb = 0; kb < kmax; kb += 32) {
        __syncthreads();
        async_copy_b128(&Kt[jj][dk], Kh + hbase + (size_t)kb * HD, voff);
        wait_async0();
        __syncthreads();

        float svv[2][8];
        #pragma unroll
        for (int nt = 0; nt < 2; ++nt) {
            v16h b0, b1;
            int krow = nt * 16 + lane15;
            #pragma unroll
            for (int r = 0; r < 8; ++r) {
                int dd = 2 * r + (laneHi ? 16 : 0);
                b0[2*r] = Kt[krow][dd];      b0[2*r+1] = Kt[krow][dd + 1];
                b1[2*r] = Kt[krow][32 + dd]; b1[2*r+1] = Kt[krow][32 + dd + 1];
            }
            v8f z = {};
            v8f s = wmma16(qa0, b0, z);
            s = wmma16(qa1, b1, s);
            int key = kb + nt * 16 + lane15;
            #pragma unroll
            for (int r = 0; r < 8; ++r) {
                float sv = s[r];
                int dist = rl[r] - key;
                if (dist >= 0) { if (dist <= WIN) sv += Bh[dist]; }
                else sv = -1e30f;
                svv[nt][r] = sv;
            }
        }
        #pragma unroll
        for (int r = 0; r < 8; ++r) {
            float pm = fmaxf(svv[0][r], svv[1][r]);
            pm = fmaxf(pm, __shfl_xor(pm, 1));
            pm = fmaxf(pm, __shfl_xor(pm, 2));
            pm = fmaxf(pm, __shfl_xor(pm, 4));
            pm = fmaxf(pm, __shfl_xor(pm, 8));
            float mn = fmaxf(mrow[r], pm);
            float pe = expf(svv[0][r] - mn) + expf(svv[1][r] - mn);
            pe += __shfl_xor(pe, 1); pe += __shfl_xor(pe, 2);
            pe += __shfl_xor(pe, 4); pe += __shfl_xor(pe, 8);
            zrow[r] = zrow[r] * expf(mrow[r] - mn) + pe;
            mrow[r] = mn;
        }
    }

    float rz[8], ti[8];
    #pragma unroll
    for (int r = 0; r < 8; ++r) {
        rz[r] = 1.0f / zrow[r];
        ti[r] = tauh / (float)(rl[r] + 1);
    }

    v8f o[4] = {};

    // ---------------- pass 2: elastic probs + PV ----------------
    for (int kb = 0; kb < kmax; kb += 32) {
        __syncthreads();
        async_copy_b128(&Kt[jj][dk], Kh + hbase + (size_t)kb * HD, voff);
        async_copy_b128(&Vt[jj][dk], Vh + hbase + (size_t)kb * HD, voff);
        wait_async0();
        __syncthreads();

        float svv[2][8];
        #pragma unroll
        for (int nt = 0; nt < 2; ++nt) {
            v16h b0, b1;
            int krow = nt * 16 + lane15;
            #pragma unroll
            for (int r = 0; r < 8; ++r) {
                int dd = 2 * r + (laneHi ? 16 : 0);
                b0[2*r] = Kt[krow][dd];      b0[2*r+1] = Kt[krow][dd + 1];
                b1[2*r] = Kt[krow][32 + dd]; b1[2*r+1] = Kt[krow][32 + dd + 1];
            }
            v8f z = {};
            v8f s = wmma16(qa0, b0, z);
            s = wmma16(qa1, b1, s);
            int key = kb + nt * 16 + lane15;
            #pragma unroll
            for (int r = 0; r < 8; ++r) {
                float sv = s[r];
                int dist = rl[r] - key;
                if (dist >= 0) { if (dist <= WIN) sv += Bh[dist]; }
                else sv = -1e30f;
                svv[nt][r] = sv;
            }
        }
        // elastic probs -> wave-private LDS (C-layout write, A-layout read)
        #pragma unroll
        for (int nt = 0; nt < 2; ++nt)
            #pragma unroll
            for (int r = 0; r < 8; ++r) {
                float p = expf(svv[nt][r] - mrow[r]) * rz[r];
                float e = fmaxf(p + ti[r], 0.0f);
                Pt[wave][r + laneHi * 8][nt * 16 + lane15] = (_Float16)e;
            }
        __builtin_amdgcn_wave_barrier();
        asm volatile("" ::: "memory");     // LDS is in-order per wave in HW

        v16h pa;
        #pragma unroll
        for (int r = 0; r < 8; ++r) {
            int kk = 2 * r + (laneHi ? 16 : 0);
            pa[2*r]   = Pt[wave][lane15][kk];
            pa[2*r+1] = Pt[wave][lane15][kk + 1];
        }
        #pragma unroll
        for (int nt = 0; nt < 4; ++nt) {
            v16h vb;
            int d = nt * 16 + lane15;
            #pragma unroll
            for (int r = 0; r < 8; ++r) {
                int j0 = 2 * r + (laneHi ? 16 : 0);
                vb[2*r]   = Vt[j0][d];
                vb[2*r+1] = Vt[j0 + 1][d];
            }
            o[nt] = wmma16(pa, vb, o[nt]);
        }
    }

    // write O (f16) laid out [B, L, H*hd] for the output projection GEMM
    #pragma unroll
    for (int nt = 0; nt < 4; ++nt) {
        int col = nt * 16 + lane15;
        #pragma unroll
        for (int r = 0; r < 8; ++r) {
            Obuf[((size_t)(bIdx * LSEQ + rl[r])) * DM + h * HD + col] = (_Float16)o[nt][r];
        }
    }
}

// ---------------------------------------------------------------------------
// Kernel 3: out = Ocat(f16) @ Wo^T  -> f32. grid (BLROWS/128, DM/64).
// O tile staged via async-to-LDS; Wo staged with f32->f16 cvt at stage time.
// ---------------------------------------------------------------------------
__global__ __launch_bounds__(256) void out_proj_kernel(
    const _Float16* __restrict__ A, const float* __restrict__ Wo,
    float* __restrict__ out)
{
    const int n0       = blockIdx.y * 64;
    const int rowBase0 = blockIdx.x * 128;
    const int tid      = threadIdx.x;
    const int lane     = tid & 31;
    const int wave     = tid >> 5;
    const int lane15   = lane & 15;
    const int laneHi   = lane >> 4;

    __shared__ _Float16 As[128][40];
    __shared__ _Float16 Wh[64][40];

    v8f acc[4] = {};

    for (int kk0 = 0; kk0 < DM; kk0 += 32) {
        __syncthreads();
        // async stage A tile: 512 B128 slots (2 per thread), already f16
        #pragma unroll
        for (int j = 0; j < 2; ++j) {
            int idx = tid + j * 256;
            int row = idx >> 2, seg = idx & 3;
            unsigned voff = (unsigned)((((size_t)(rowBase0 + row)) * DM + seg * 8) * sizeof(_Float16));
            async_copy_b128(&As[row][seg * 8], A + kk0, voff);
        }
        // stage Wo tile with cvt
        #pragma unroll
        for (int j = 0; j < 2; ++j) {
            int idx = tid + j * 256;
            int wr = idx >> 3, c4 = idx & 7;
            float4 v = *(const float4*)(Wo + (size_t)(n0 + wr) * DM + kk0 + c4 * 4);
            Wh[wr][c4*4+0] = (_Float16)v.x; Wh[wr][c4*4+1] = (_Float16)v.y;
            Wh[wr][c4*4+2] = (_Float16)v.z; Wh[wr][c4*4+3] = (_Float16)v.w;
        }
        wait_async0();
        __syncthreads();

        v16h af;
        int arow = wave * 16 + lane15;
        #pragma unroll
        for (int r = 0; r < 8; ++r) {
            int kk = 2 * r + (laneHi ? 16 : 0);
            af[2*r]   = As[arow][kk];
            af[2*r+1] = As[arow][kk + 1];
        }
        #pragma unroll
        for (int nt = 0; nt < 4; ++nt) {
            v16h bf;
            int n = nt * 16 + lane15;
            #pragma unroll
            for (int r = 0; r < 8; ++r) {
                int kk = 2 * r + (laneHi ? 16 : 0);
                bf[2*r]   = Wh[n][kk];
                bf[2*r+1] = Wh[n][kk + 1];
            }
            acc[nt] = wmma16(af, bf, acc[nt]);
        }
    }

    #pragma unroll
    for (int nt = 0; nt < 4; ++nt)
        #pragma unroll
        for (int r = 0; r < 8; ++r) {
            out[(size_t)(rowBase0 + wave * 16 + r + laneHi * 8) * DM + n0 + nt * 16 + lane15] =
                acc[nt][r];
        }
}

// ---------------------------------------------------------------------------
extern "C" void kernel_launch(void* const* d_in, const int* in_sizes, int n_in,
                              void* d_out, int out_size, void* d_ws, size_t ws_size,
                              hipStream_t stream) {
    (void)in_sizes; (void)n_in; (void)out_size; (void)ws_size;
    const float* x      = (const float*)d_in[0];
    const float* Wq     = (const float*)d_in[1];
    const float* Wk     = (const float*)d_in[2];
    const float* Wv     = (const float*)d_in[3];
    const float* Wo     = (const float*)d_in[4];
    const float* biases = (const float*)d_in[5];
    const float* tau    = (const float*)d_in[6];
    float* out = (float*)d_out;

    const size_t elems = (size_t)2 * H_NUM * LSEQ * HD;   // 4,194,304 halfs = 8MB
    _Float16* Qh   = (_Float16*)d_ws;
    _Float16* Kh   = Qh + elems;
    _Float16* Vh   = Kh + elems;
    _Float16* Obuf = Vh + elems;                          // B*L*D halfs = 8MB

    dim3 g1(BLROWS / 128, H_NUM);
    proj_rope_kernel<<<g1, 256, 0, stream>>>(x, Wq, Wk, Wv, Qh, Kh, Vh);

    dim3 g2(LSEQ / 128, H_NUM, 2);
    attn_kernel<<<g2, 256, 0, stream>>>(Qh, Kh, Vh, biases, tau, Obuf);

    dim3 g3(BLROWS / 128, DM / 64);
    out_proj_kernel<<<g3, 256, 0, stream>>>(Obuf, Wo, out);
}
